// Model_39676907887723
// MI455X (gfx1250) — compile-verified, matrix operand order checked by reading, not run on hardware
//
#include <hip/hip_runtime.h>

typedef __attribute__((ext_vector_type(16))) _Float16 v16h;
typedef __attribute__((ext_vector_type(8)))  _Float16 v8h;
typedef __attribute__((ext_vector_type(4)))  _Float16 v4h;
typedef __attribute__((ext_vector_type(8)))  float    v8f;
typedef __attribute__((ext_vector_type(4)))  float    v4f;

#define SEQ   2048
#define DHEAD 64
#define QTILE 16                 // q rows per wave
#define WAVES 8                  // waves per block (256 threads, wave32)
#define QBLK  (QTILE * WAVES)    // 128 q rows per block
#define KTILE 64                 // keys per iteration (amortizes per-row softmax state)
#define NKIT  (SEQ / KTILE)      // 32 iterations
#define KPAD  (DHEAD + 8)        // 72 halfs = 144B row stride (16B aligned)

union U16 { v16h v; v8h h[2]; };

// ---- 16-lane butterfly max on the VALU via DPP16 (no LDS pipe) ----
template <int CTRL>
__device__ __forceinline__ float dpp_f32(float x) {
    int i = __builtin_bit_cast(int, x);
    i = __builtin_amdgcn_update_dpp(0, i, CTRL, 0xF, 0xF, true);
    return __builtin_bit_cast(float, i);
}
__device__ __forceinline__ float red16_max(float v) {
    v = fmaxf(v, dpp_f32<0xB1>(v));    // quad_perm(1,0,3,2) : xor 1
    v = fmaxf(v, dpp_f32<0x4E>(v));    // quad_perm(2,3,0,1) : xor 2
    v = fmaxf(v, dpp_f32<0x141>(v));   // row_half_mirror    : 8-lane mirror
    v = fmaxf(v, dpp_f32<0x140>(v));   // row_mirror         : 16-lane mirror
    return v;
}

__global__ __launch_bounds__(256, 1)
void fa_fwd_kernel(const float* __restrict__ Q, const float* __restrict__ K,
                   const float* __restrict__ V, const float* __restrict__ M,
                   float* __restrict__ O)
{
    // double-buffered K tile: row = key (0..63), col = d (0..63)
    __shared__ __attribute__((aligned(16))) _Float16 sK [2][KTILE][KPAD];
    // double-buffered V^T tile: row = d (0..63), col = key (0..63)
    __shared__ __attribute__((aligned(16))) _Float16 sVT[2][DHEAD][KTILE + 8];
    // per-wave P staging: row = q (0..15), col = key (0..63)
    __shared__ __attribute__((aligned(16))) _Float16 sP [WAVES][QTILE][KTILE + 8];

    const int tid  = threadIdx.x;
    const int lane = tid & 31;
    const int wave = tid >> 5;
    const int half = lane >> 4;   // 16-lane half of the wave
    const int ln   = lane & 15;   // M for A-frags / N for B-frags and C cols

    const int bh      = blockIdx.x / (SEQ / QBLK);
    const int qblk    = blockIdx.x % (SEQ / QBLK);
    const size_t base = (size_t)bh * SEQ * DHEAD;
    const int qtile0  = qblk * QBLK + wave * QTILE;

    // staging: 64x64 f32 tile = 1024 float4 slots; thread t owns slots t+256*i
    const int ld_row = tid >> 4;              // key rows: +16 per slot group
    const int ld_c4  = (tid & 15) << 2;       // d col base

    // ---- Q tile -> two A-matrix fragments (16x32 f16, K-dim = head dim) ----
    U16 qa[2];
    {
        const float* qp = Q + base + (size_t)(qtile0 + ln) * DHEAD;
        #pragma unroll
        for (int f = 0; f < 2; ++f) {
            #pragma unroll
            for (int g = 0; g < 2; ++g) {
                const int d0 = f * 32 + g * 16 + half * 8;
                v4f x0 = *(const v4f*)(qp + d0);
                v4f x1 = *(const v4f*)(qp + d0 + 4);
                v8h h;
                #pragma unroll
                for (int j = 0; j < 4; ++j) {
                    h[j]     = (_Float16)x0[j];
                    h[4 + j] = (_Float16)x1[j];
                }
                qa[f].h[g] = h;
            }
        }
    }

    // all-ones B fragment: wmma(P, ones) gives row-sums of P in C layout
    U16 onesb;
    #pragma unroll
    for (int e = 0; e < 16; ++e) onesb.v[e] = (_Float16)1.0f;

    // online-softmax state in C layout (row r of C -> q row r + 8*half)
    float mrow[8];
    v8f acc[4], lacc;
    const v8f vzero = {};
    #pragma unroll
    for (int r = 0; r < 8; ++r) mrow[r] = -3.0e38f;
    #pragma unroll
    for (int c = 0; c < 4; ++c) acc[c] = vzero;
    lacc = vzero;

    const float* Kbh = K + base;
    const float* Vbh = V + base;

    v4f kreg[4], vreg[4];
    // ---- prologue: stage tile 0 into buffer 0 ----
    #pragma unroll
    for (int i = 0; i < 4; ++i) {
        kreg[i] = *(const v4f*)(Kbh + (ld_row + i * 16) * DHEAD + ld_c4);
        vreg[i] = *(const v4f*)(Vbh + (ld_row + i * 16) * DHEAD + ld_c4);
    }
    #pragma unroll
    for (int i = 0; i < 4; ++i) {
        v4h kp;
        #pragma unroll
        for (int j = 0; j < 4; ++j) {
            kp[j] = (_Float16)kreg[i][j];
            sVT[0][ld_c4 + j][ld_row + i * 16] = (_Float16)vreg[i][j];
        }
        *(v4h*)&sK[0][ld_row + i * 16][ld_c4] = kp;   // ds_store_b64
    }
    __syncthreads();

    for (int kt = 0; kt < NKIT; ++kt) {
        const int cur = kt & 1;

        // ---- issue next tile's global loads (hidden behind the WMMA block) ----
        if (kt + 1 < NKIT) {
            const float* Kp = Kbh + (size_t)(kt + 1) * KTILE * DHEAD;
            const float* Vp = Vbh + (size_t)(kt + 1) * KTILE * DHEAD;
            #pragma unroll
            for (int i = 0; i < 4; ++i) {
                kreg[i] = *(const v4f*)(Kp + (ld_row + i * 16) * DHEAD + ld_c4);
                vreg[i] = *(const v4f*)(Vp + (ld_row + i * 16) * DHEAD + ld_c4);
            }
        }
        if (kt + 2 < NKIT) {   // prime L2 one more tile out (global_prefetch_b8)
            __builtin_prefetch(Kbh + (size_t)(kt + 2) * KTILE * DHEAD + tid * 16, 0, 1);
            __builtin_prefetch(Vbh + (size_t)(kt + 2) * KTILE * DHEAD + tid * 16, 0, 1);
        }

        // ---- S = Q K^T : four 16x16 f32 tiles (keys t*16..t*16+15), K-dim 64 ----
        v8f s[4];
        #pragma unroll
        for (int t = 0; t < 4; ++t) s[t] = vzero;
        #pragma unroll
        for (int f = 0; f < 2; ++f) {
            #pragma unroll
            for (int t = 0; t < 4; ++t) {
                U16 kb;
                #pragma unroll
                for (int g = 0; g < 2; ++g)
                    kb.h[g] = *(const v8h*)&sK[cur][t * 16 + ln][f * 32 + g * 16 + half * 8];
                s[t] = __builtin_amdgcn_wmma_f32_16x16x32_f16(false, qa[f].v, false, kb.v,
                                                              (short)0, s[t], false, false);
            }
        }

        // ---- + mask, online softmax (scale = 1/sqrt(1) = 1) ----
        const float* mp = M + (size_t)(qtile0 + half * 8) * SEQ + kt * KTILE + ln;
        _Float16 ph[4][8];
        #pragma unroll
        for (int r = 0; r < 8; ++r) {
            const float* mq = mp + (size_t)r * SEQ;   // mask is L2-resident (16MB << 192MB)
            float a[4];
            #pragma unroll
            for (int t = 0; t < 4; ++t) a[t] = s[t][r] + mq[t * 16];
            const float rmax = red16_max(fmaxf(fmaxf(a[0], a[1]), fmaxf(a[2], a[3])));
            const float mnew = fmaxf(mrow[r], rmax);
            const float corr = __expf(mrow[r] - mnew);
            mrow[r] = mnew;
            #pragma unroll
            for (int t = 0; t < 4; ++t) ph[t][r] = (_Float16)__expf(a[t] - mnew);
            #pragma unroll
            for (int c = 0; c < 4; ++c) acc[c][r] *= corr;
            lacc[r] *= corr;
        }

        // ---- transpose P: C layout -> A layout via wave-private LDS (in-order DS) ----
        #pragma unroll
        for (int r = 0; r < 8; ++r)
            #pragma unroll
            for (int t = 0; t < 4; ++t)
                sP[wave][r + half * 8][t * 16 + ln] = ph[t][r];
        U16 pa[2];
        #pragma unroll
        for (int sfr = 0; sfr < 2; ++sfr)
            #pragma unroll
            for (int g = 0; g < 2; ++g)
                pa[sfr].h[g] = *(const v8h*)&sP[wave][ln][sfr * 32 + g * 16 + half * 8];

        // ---- O += P V (4 d-chunks, K-dim 64) ; l += P . 1 via WMMA ----
        #pragma unroll
        for (int sfr = 0; sfr < 2; ++sfr) {
            #pragma unroll
            for (int c = 0; c < 4; ++c) {
                U16 vb;
                #pragma unroll
                for (int g = 0; g < 2; ++g)
                    vb.h[g] = *(const v8h*)&sVT[cur][c * 16 + ln][sfr * 32 + g * 16 + half * 8];
                acc[c] = __builtin_amdgcn_wmma_f32_16x16x32_f16(false, pa[sfr].v, false, vb.v,
                                                                (short)0, acc[c], false, false);
            }
            lacc = __builtin_amdgcn_wmma_f32_16x16x32_f16(false, pa[sfr].v, false, onesb.v,
                                                          (short)0, lacc, false, false);
        }

        // ---- commit next tile into the other buffer; single barrier per iter ----
        if (kt + 1 < NKIT) {
            #pragma unroll
            for (int i = 0; i < 4; ++i) {
                v4h kp;
                #pragma unroll
                for (int j = 0; j < 4; ++j) {
                    kp[j] = (_Float16)kreg[i][j];
                    sVT[cur ^ 1][ld_c4 + j][ld_row + i * 16] = (_Float16)vreg[i][j];
                }
                *(v4h*)&sK[cur ^ 1][ld_row + i * 16][ld_c4] = kp;   // ds_store_b64
            }
        }
        __syncthreads();
    }

    // ---- normalize and write O (coalesced over 16 lanes) ----
    #pragma unroll
    for (int r = 0; r < 8; ++r) {
        const float inv = 1.0f / lacc[r];   // every column of lacc holds the row sum
        float* op = O + base + (size_t)(qtile0 + r + half * 8) * DHEAD + ln;
        #pragma unroll
        for (int c = 0; c < 4; ++c)
            op[c * 16] = acc[c][r] * inv;
    }
}

extern "C" void kernel_launch(void* const* d_in, const int* in_sizes, int n_in,
                              void* d_out, int out_size, void* d_ws, size_t ws_size,
                              hipStream_t stream) {
    (void)in_sizes; (void)n_in; (void)out_size; (void)d_ws; (void)ws_size;
    const float* q = (const float*)d_in[0];
    const float* k = (const float*)d_in[1];
    const float* v = (const float*)d_in[2];
    const float* m = (const float*)d_in[3];
    float* o = (float*)d_out;
    const int blocks = 4 * 16 * (SEQ / QBLK);   // B * H * q-blocks = 1024
    fa_fwd_kernel<<<blocks, 256, 0, stream>>>(q, k, v, m, o);
}
